// AttentionWithMemory_27358941675954
// MI455X (gfx1250) — matrix operand order repaired and checked by reading, not compiled, not run on hardware
//
#include <hip/hip_runtime.h>
#include <hip/hip_bf16.h>

#define SEQ   1024
#define DIMSZ 1024
#define HEADS 16
#define HDIM  64
#define BATCH 8
#define BSROWS (BATCH * SEQ)       // 8192
#define D3    (3 * DIMSZ)          // 3072

typedef __attribute__((ext_vector_type(16))) __bf16 v16bf;
typedef __attribute__((ext_vector_type(8)))  float  v8f;

// ---- WMMA fragment helpers (16x16x32 bf16, contiguous K order shared by A and B) ----
// A (16x32, bf16): lane L holds row M=L&15; khalf=L>>4 selects K half:
//   elem i -> K = kbase + khalf*16 + i
static __device__ __forceinline__ v16bf frag_a_bf16(const __bf16* row, int kbase, int lane) {
    const __bf16* p = row + kbase + ((lane >> 4) << 4);
    v16bf a;
#pragma unroll
    for (int i = 0; i < 8; ++i) a[i] = p[i];
#pragma unroll
    for (int i = 0; i < 8; ++i) a[8 + i] = p[8 + i];
    return a;
}

static __device__ __forceinline__ v16bf frag_a_f32(const float* row, int kbase, int lane) {
    const float* p = row + kbase + ((lane >> 4) << 4);
    v16bf a;
#pragma unroll
    for (int i = 0; i < 8; ++i) a[i] = (__bf16)p[i];
#pragma unroll
    for (int i = 0; i < 8; ++i) a[8 + i] = (__bf16)p[8 + i];
    return a;
}

static __device__ __forceinline__ v8f wmma_bf16(v16bf a, v16bf b, v8f c) {
    return __builtin_amdgcn_wmma_f32_16x16x32_bf16(false, a, false, b, (short)0, c,
                                                   false, false);
}

// ---- fp32 -> bf16 cast kernel ----
__global__ void __launch_bounds__(256)
cast_f32_bf16(const float* __restrict__ in, __bf16* __restrict__ out, int n) {
    int i = blockIdx.x * 256 + threadIdx.x;
    if (i < n) out[i] = (__bf16)in[i];
}

// ---- GEMM:  C[M,N] = A[M,K] * W[N,K]^T + bias[N]   (bf16 in, f32 out) ----
// One wave computes a 16x128 strip (8 accumulators, A fragment reused 8x).
// Block = 4 waves = 64 rows x 128 cols.
__global__ void __launch_bounds__(128)
gemm_nt_bias(const __bf16* __restrict__ A, const __bf16* __restrict__ W,
             const float* __restrict__ bias, float* __restrict__ C,
             int K, int N) {
    const int lane = threadIdx.x & 31;
    const int wave = threadIdx.x >> 5;
    const int m0 = blockIdx.y * 64 + wave * 16;
    const int n0 = blockIdx.x * 128;
    const int lm = lane & 15;
    const int kh16 = (lane >> 4) << 4;

    v8f zero = {};
    v8f acc[8];
#pragma unroll
    for (int t = 0; t < 8; ++t) acc[t] = zero;

    const __bf16* arow = A + (size_t)(m0 + lm) * K;

    for (int kb = 0; kb < K; kb += 32) {
        __builtin_prefetch(arow + kb + 64, 0, 0);   // global_prefetch_b8
        v16bf a = frag_a_bf16(arow, kb, lane);
#pragma unroll
        for (int t = 0; t < 8; ++t) {
            const __bf16* wp = W + (size_t)(n0 + t * 16 + lm) * K + kb + kh16;
            v16bf b;
#pragma unroll
            for (int i = 0; i < 16; ++i) b[i] = wp[i];
            acc[t] = wmma_bf16(a, b, acc[t]);
        }
    }

    const int mrow = m0 + ((lane >> 4) << 3);
#pragma unroll
    for (int t = 0; t < 8; ++t) {
        const int col = n0 + t * 16 + lm;
        const float bv = bias[col];
#pragma unroll
        for (int r = 0; r < 8; ++r)
            C[(size_t)(mrow + r) * N + col] = acc[t][r] + bv;
    }
}

// ---- Local-window attention: one wave per (batch, head, 16-query tile) ----
// Band of 48 keys [q0-16, q0+31]; mask |key-qpos|<=16 applied analytically.
__global__ void __launch_bounds__(32)
local_attn(const float* __restrict__ qkv, float* __restrict__ ctx) {
    __shared__ float pLds[16][64];     // probs, padded K 48..63 = 0

    const int lane = threadIdx.x;
    const int qt = blockIdx.x;         // 0..63
    const int h  = blockIdx.y;         // 0..15
    const int b  = blockIdx.z;         // 0..7
    const int q0 = qt * 16;
    const int kstart = q0 - 16;
    const int lm = lane & 15;
    const int khalf = lane >> 4;

    const float* base = qkv + (size_t)b * SEQ * D3;
    const float* Qp = base + h * HDIM;             // row stride D3
    const float* Kp = base + DIMSZ + h * HDIM;
    const float* Vp = base + 2 * DIMSZ + h * HDIM;

    // Q fragments (16 rows x 64 k, two K-steps)
    const float* qrow = Qp + (size_t)(q0 + lm) * D3;
    v16bf qa0 = frag_a_f32(qrow, 0, lane);
    v16bf qa1 = frag_a_f32(qrow, 32, lane);

    v8f zero = {};
    v8f s[3] = {zero, zero, zero};
#pragma unroll
    for (int t = 0; t < 3; ++t) {
        int key = kstart + t * 16 + lm;
        int kc = min(max(key, 0), SEQ - 1);
        const float* kr = Kp + (size_t)kc * D3 + (khalf << 4);
        v16bf b0, b1;
#pragma unroll
        for (int i = 0; i < 16; ++i) b0[i] = (__bf16)kr[i];
#pragma unroll
        for (int i = 0; i < 16; ++i) b1[i] = (__bf16)kr[32 + i];
        s[t] = wmma_bf16(qa0, b0, s[t]);
        s[t] = wmma_bf16(qa1, b1, s[t]);
    }

    // masked softmax over the 48-wide band (rows live in 16-lane halves)
    const float scale = 0.125f;        // 1/sqrt(64)
    float p[3][8];
#pragma unroll
    for (int r = 0; r < 8; ++r) {
        const int qpos = q0 + khalf * 8 + r;
        float mx = -3.0e38f;
#pragma unroll
        for (int t = 0; t < 3; ++t) {
            int key = kstart + t * 16 + lm;
            bool ok = (key >= 0) && (key < SEQ) &&
                      (key >= qpos - 16) && (key <= qpos + 16);
            float v = ok ? s[t][r] * scale : -3.0e38f;
            p[t][r] = v;
            mx = fmaxf(mx, v);
        }
#pragma unroll
        for (int off = 1; off < 16; off <<= 1)
            mx = fmaxf(mx, __shfl_xor(mx, off));   // stays within 16-lane half
        float sum = 0.f;
#pragma unroll
        for (int t = 0; t < 3; ++t) {
            float e = (p[t][r] <= -1.0e37f) ? 0.f : __expf(p[t][r] - mx);
            p[t][r] = e;
            sum += e;
        }
#pragma unroll
        for (int off = 1; off < 16; off <<= 1)
            sum += __shfl_xor(sum, off);
        float inv = 1.f / sum;                      // window always non-empty
#pragma unroll
        for (int t = 0; t < 3; ++t) p[t][r] *= inv;
    }

    // re-shape probs C-layout -> A-layout via LDS (pad cols 48..63 with 0)
    for (int idx = lane; idx < 16 * 16; idx += 32)
        pLds[idx >> 4][48 + (idx & 15)] = 0.f;
#pragma unroll
    for (int t = 0; t < 3; ++t)
#pragma unroll
        for (int r = 0; r < 8; ++r)
            pLds[khalf * 8 + r][t * 16 + lm] = p[t][r];
    __syncthreads();

    v16bf pa0 = frag_a_f32(&pLds[lm][0], 0, lane);
    v16bf pa1 = frag_a_f32(&pLds[lm][0], 32, lane);

    v8f o[4] = {zero, zero, zero, zero};
#pragma unroll
    for (int t2 = 0; t2 < 4; ++t2) {
        const int ncol = t2 * 16 + lm;
        v16bf vb0, vb1;
#pragma unroll
        for (int i = 0; i < 16; ++i) {
            int key0 = kstart + (khalf << 4) + i;
            int kc0 = min(max(key0, 0), SEQ - 1);
            vb0[i] = (__bf16)Vp[(size_t)kc0 * D3 + ncol];
            int key1 = kstart + 32 + (khalf << 4) + i;      // 48..63 hit P==0
            int kc1 = min(max(key1, 0), SEQ - 1);
            vb1[i] = (__bf16)Vp[(size_t)kc1 * D3 + ncol];
        }
        o[t2] = wmma_bf16(pa0, vb0, o[t2]);
        o[t2] = wmma_bf16(pa1, vb1, o[t2]);
    }

    float* crow = ctx + (size_t)b * SEQ * DIMSZ + (size_t)q0 * DIMSZ + h * HDIM;
#pragma unroll
    for (int t2 = 0; t2 < 4; ++t2)
#pragma unroll
        for (int r = 0; r < 8; ++r)
            crow[(size_t)(khalf * 8 + r) * DIMSZ + t2 * 16 + lm] = o[t2][r];
}

// ---------------------------------------------------------------------------
extern "C" void kernel_launch(void* const* d_in, const int* in_sizes, int n_in,
                              void* d_out, int out_size, void* d_ws, size_t ws_size,
                              hipStream_t stream) {
    const float* x     = (const float*)d_in[0];   // [8, 1024, 1024]
    const float* w_in  = (const float*)d_in[1];   // [3072, 1024]
    const float* b_in  = (const float*)d_in[2];   // [3072]
    const float* w_out = (const float*)d_in[3];   // [1024, 1024]
    const float* b_out = (const float*)d_in[4];   // [1024]
    // d_in[5] = mask: computed analytically on device, not read
    float* out = (float*)d_out;                   // [8, 1024, 1024]

    // workspace carve (all sizes 4KB-multiple aligned)
    char* ws = (char*)d_ws;
    __bf16* xb   = (__bf16*)ws; ws += (size_t)BSROWS * DIMSZ * sizeof(__bf16);
    __bf16* wib  = (__bf16*)ws; ws += (size_t)D3 * DIMSZ * sizeof(__bf16);
    __bf16* wob  = (__bf16*)ws; ws += (size_t)DIMSZ * DIMSZ * sizeof(__bf16);
    float*  qkv  = (float*)ws;  ws += (size_t)BSROWS * D3 * sizeof(float);
    float*  ctx  = (float*)ws;  ws += (size_t)BSROWS * DIMSZ * sizeof(float);
    __bf16* ctxb = (__bf16*)ws;

    const int nX  = BSROWS * DIMSZ;   // 8388608
    const int nWi = D3 * DIMSZ;       // 3145728
    const int nWo = DIMSZ * DIMSZ;    // 1048576

    cast_f32_bf16<<<(nX  + 255) / 256, 256, 0, stream>>>(x, xb, nX);
    cast_f32_bf16<<<(nWi + 255) / 256, 256, 0, stream>>>(w_in, wib, nWi);
    cast_f32_bf16<<<(nWo + 255) / 256, 256, 0, stream>>>(w_out, wob, nWo);

    // qkv = x @ w_in^T + b_in   : M=8192, K=1024, N=3072
    gemm_nt_bias<<<dim3(D3 / 128, BSROWS / 64), 128, 0, stream>>>(
        xb, wib, b_in, qkv, DIMSZ, D3);

    // banded attention -> ctx
    local_attn<<<dim3(SEQ / 16, HEADS, BATCH), 32, 0, stream>>>(qkv, ctx);

    cast_f32_bf16<<<(nX + 255) / 256, 256, 0, stream>>>(ctx, ctxb, nX);

    // out = ctx @ w_out^T + b_out : M=8192, K=1024, N=1024
    gemm_nt_bias<<<dim3(DIMSZ / 128, BSROWS / 64), 128, 0, stream>>>(
        ctxb, wob, b_out, out, DIMSZ, DIMSZ);
}
